// TruncatedMLP_71863392796798
// MI455X (gfx1250) — compile-verified, matrix operand order checked by reading, not run on hardware
//
#include <hip/hip_runtime.h>
#include <hip/hip_bf16.h>

// ---------------------------------------------------------------------------
// Fused GNN edge-MLP for MI455X (gfx1250, wave32, WMMA).
//   out = LN( silu( [efeat | src_g | dst_g] @ W1stackT + b1 ) @ W2T + b2 )
// GEMMs run on v_wmma_f32_16x16x32_bf16 with a 3-term hi/lo bf16 split
// (~fp32 accuracy). Weights are pre-split into register-layout fragments.
// ---------------------------------------------------------------------------

typedef __attribute__((ext_vector_type(16))) __bf16 v16bf;
typedef __attribute__((ext_vector_type(8)))  float  v8f;

#define HID 128
#define NT_TILES 8      // 128 / 16 output-column tiles
#define KT1 12          // 384 / 32 k-tiles for GEMM1 (concat K)
#define KT2 4           // 128 / 32 k-tiles for GEMM2
#define FRAG_BYTES 1024 // 32 lanes * 32 bytes (v16bf) per fragment
#define WS_B2_OFF (KT1 * NT_TILES * 2 * FRAG_BYTES) // 196608
#define LDS_STRIDE 132  // 128 + 4 pad -> rows land on distinct banks
#define WAVES_PER_BLOCK 4

// K offset inside a 16x32 bf16 A tile for VGPR pair v (0..7), per ISA 7.12.2:
// lanes 0-15: V0..3 -> K={0,1},{2,3},{4,5},{6,7}; V4..7 -> K={16..23} pairs
// lanes 16-31: +8. (B 32x16 mirrored with lane = N.)
__device__ __forceinline__ int a_koff(int v, int laneHi) {
  int base = (v < 4) ? (v << 1) : (16 + ((v - 4) << 1));
  return base + (laneHi ? 8 : 0);
}

__device__ __forceinline__ void split_bf16(float x, __bf16& hi, __bf16& lo) {
  hi = (__bf16)x;
  lo = (__bf16)(x - (float)hi);
}

// ---------------------------------------------------------------------------
// Kernel 1: pre-split weights into WMMA B-fragment layout (bf16 hi/lo).
// One wave per fragment-pair. f<96 -> stacked W1 (K=384), else W2 (K=128).
// ---------------------------------------------------------------------------
__global__ __launch_bounds__(32) void prep_weights_kernel(
    const float* __restrict__ w_ef, const float* __restrict__ w_src,
    const float* __restrict__ w_dst, const float* __restrict__ w2,
    char* __restrict__ ws) {
  const int f = blockIdx.x;      // 0..127
  const int lane = threadIdx.x & 31;
  const int laneHi = lane >> 4;
  const int n16 = lane & 15;

  int kt, nt, isW2;
  char* outb;
  if (f < KT1 * NT_TILES) { kt = f >> 3; nt = f & 7; isW2 = 0; outb = ws; }
  else { int g = f - KT1 * NT_TILES; kt = g >> 3; nt = g & 7; isW2 = 1; outb = ws + WS_B2_OFF; }

  const int nrow = nt * 16 + n16;  // output-feature row of W (B column)
  v16bf hi, lo;
#pragma unroll
  for (int v = 0; v < 8; ++v) {
    const int kg = kt * 32 + a_koff(v, laneHi);
    const float* W;
    int kl;
    if (isW2) { W = w2; kl = kg; }
    else {
      const int region = kg >> 7;              // 0:efeat 1:src 2:dst
      W = (region == 0) ? w_ef : (region == 1) ? w_src : w_dst;
      kl = kg & 127;
    }
    const float2 x = *(const float2*)(W + (size_t)nrow * HID + kl);
    __bf16 h0, l0, h1, l1;
    split_bf16(x.x, h0, l0);
    split_bf16(x.y, h1, l1);
    hi[2 * v] = h0; hi[2 * v + 1] = h1;
    lo[2 * v] = l0; lo[2 * v + 1] = l1;
  }
  char* p = outb + (size_t)((kt * NT_TILES + nt) * 2) * FRAG_BYTES + (size_t)lane * 32;
  *(v16bf*)p = hi;
  *(v16bf*)(p + FRAG_BYTES) = lo;
}

// ---------------------------------------------------------------------------
// Kernel 2: fused per-edge pipeline. One wave32 owns 16 edges.
// ---------------------------------------------------------------------------
__global__ __launch_bounds__(WAVES_PER_BLOCK * 32) void fused_edge_kernel(
    const float* __restrict__ efeat, const float* __restrict__ src_feat,
    const float* __restrict__ dst_feat, const int* __restrict__ src_idx,
    const int* __restrict__ dst_idx, const float* __restrict__ b1,
    const float* __restrict__ b2, const float* __restrict__ ln_g,
    const float* __restrict__ ln_b, const char* __restrict__ ws,
    float* __restrict__ out, long long E) {
  __shared__ float lds[WAVES_PER_BLOCK * 16 * LDS_STRIDE];

  const int lane = threadIdx.x & 31;
  const int wv = threadIdx.x >> 5;
  const long long blk = (long long)blockIdx.x * WAVES_PER_BLOCK + wv;
  const long long e0 = blk * 16;
  if (e0 >= E) return;                         // wave-uniform: EXEC stays all-1s

  const int laneHi = lane >> 4;
  const int n16 = lane & 15;

  long long erow = e0 + n16;
  if (erow >= E) erow = E - 1;                 // clamp reads for the tail tile
  const float* efp = efeat + erow * HID;
  const float* sp = src_feat + (long long)src_idx[erow] * HID;  // L2-resident gather
  const float* dp = dst_feat + (long long)dst_idx[erow] * HID;

  // ---------------- GEMM1: K = 384 (efeat | src_g | dst_g), acc init = b1 ---
  v8f acc[NT_TILES];
#pragma unroll
  for (int nt = 0; nt < NT_TILES; ++nt) {
    const float bv = b1[nt * 16 + n16];
#pragma unroll
    for (int i = 0; i < 8; ++i) acc[nt][i] = bv;
  }

#pragma unroll
  for (int kt = 0; kt < KT1; ++kt) {
    const float* ap = (kt < 4) ? efp : (kt < 8) ? sp : dp;
    const int kb = (kt & 3) * 32;
    v16bf ah, al;
#pragma unroll
    for (int v = 0; v < 8; ++v) {
      const float2 x = *(const float2*)(ap + kb + a_koff(v, laneHi));
      __bf16 h0, l0, h1, l1;
      split_bf16(x.x, h0, l0);
      split_bf16(x.y, h1, l1);
      ah[2 * v] = h0; ah[2 * v + 1] = h1;
      al[2 * v] = l0; al[2 * v + 1] = l1;
    }
    const char* wb = ws + (size_t)(kt * NT_TILES) * 2 * FRAG_BYTES + (size_t)lane * 32;
#pragma unroll
    for (int nt = 0; nt < NT_TILES; ++nt) {
      const char* p = wb + (size_t)nt * 2 * FRAG_BYTES;
      const v16bf bh = *(const v16bf*)p;
      const v16bf bl = *(const v16bf*)(p + FRAG_BYTES);
      acc[nt] = __builtin_amdgcn_wmma_f32_16x16x32_bf16(false, ah, false, bh, (short)0, acc[nt], false, false);
      acc[nt] = __builtin_amdgcn_wmma_f32_16x16x32_bf16(false, ah, false, bl, (short)0, acc[nt], false, false);
      acc[nt] = __builtin_amdgcn_wmma_f32_16x16x32_bf16(false, al, false, bh, (short)0, acc[nt], false, false);
    }
  }

  // ---------------- SiLU, C-layout -> A-layout via wave-private LDS slab ----
  float* slab = lds + wv * 16 * LDS_STRIDE;
  const int rowOff = laneHi * 8;               // C layout: lanes>=16 hold M = r+8
#pragma unroll
  for (int nt = 0; nt < NT_TILES; ++nt) {
    const int col = nt * 16 + n16;
#pragma unroll
    for (int r = 0; r < 8; ++r) {
      const float v = acc[nt][r];
      const float s = v / (1.0f + __expf(-v));
      slab[(r + rowOff) * LDS_STRIDE + col] = s;
    }
  }
  // Same-wave DS ops complete in order; fence compiler + hardware before reads.
  asm volatile("s_wait_dscnt 0" ::: "memory");

  // ---------------- GEMM2: K = 128, acc init = b2 ---------------------------
  v8f acc2[NT_TILES];
#pragma unroll
  for (int nt = 0; nt < NT_TILES; ++nt) {
    const float bv = b2[nt * 16 + n16];
#pragma unroll
    for (int i = 0; i < 8; ++i) acc2[nt][i] = bv;
  }
  const char* ws2 = ws + WS_B2_OFF;
#pragma unroll
  for (int kt = 0; kt < KT2; ++kt) {
    v16bf ah, al;
#pragma unroll
    for (int v = 0; v < 8; ++v) {
      const int kl = kt * 32 + a_koff(v, laneHi);
      const float2 x = *(const float2*)&slab[n16 * LDS_STRIDE + kl];
      __bf16 h0, l0, h1, l1;
      split_bf16(x.x, h0, l0);
      split_bf16(x.y, h1, l1);
      ah[2 * v] = h0; ah[2 * v + 1] = h1;
      al[2 * v] = l0; al[2 * v + 1] = l1;
    }
    const char* wb = ws2 + (size_t)(kt * NT_TILES) * 2 * FRAG_BYTES + (size_t)lane * 32;
#pragma unroll
    for (int nt = 0; nt < NT_TILES; ++nt) {
      const char* p = wb + (size_t)nt * 2 * FRAG_BYTES;
      const v16bf bh = *(const v16bf*)p;
      const v16bf bl = *(const v16bf*)(p + FRAG_BYTES);
      acc2[nt] = __builtin_amdgcn_wmma_f32_16x16x32_bf16(false, ah, false, bh, (short)0, acc2[nt], false, false);
      acc2[nt] = __builtin_amdgcn_wmma_f32_16x16x32_bf16(false, ah, false, bl, (short)0, acc2[nt], false, false);
      acc2[nt] = __builtin_amdgcn_wmma_f32_16x16x32_bf16(false, al, false, bh, (short)0, acc2[nt], false, false);
    }
  }

  // ---------------- LayerNorm over 128 cols per row -------------------------
  // Row M=r(+8) lives in lanes [g*16, g*16+16); xor masks 1,2,4,8 reduce
  // each 16-lane half independently -> per-row mean / var in the right lanes.
  float rmu[8], rrstd[8];
#pragma unroll
  for (int r = 0; r < 8; ++r) {
    float s = 0.0f, sq = 0.0f;
#pragma unroll
    for (int nt = 0; nt < NT_TILES; ++nt) {
      const float v = acc2[nt][r];
      s += v; sq += v * v;
    }
#pragma unroll
    for (int m = 1; m < 16; m <<= 1) {
      s += __shfl_xor(s, m, 32);
      sq += __shfl_xor(sq, m, 32);
    }
    const float mu = s * (1.0f / 128.0f);
    const float var = sq * (1.0f / 128.0f) - mu * mu;
    rmu[r] = mu;
    rrstd[r] = rsqrtf(var + 1e-5f);
  }

#pragma unroll
  for (int nt = 0; nt < NT_TILES; ++nt) {
    const int col = nt * 16 + n16;
    const float g = ln_g[col];
    const float b = ln_b[col];
#pragma unroll
    for (int r = 0; r < 8; ++r) {
      const long long orow = e0 + r + rowOff;
      if (orow < E)
        out[orow * HID + col] = (acc2[nt][r] - rmu[r]) * rrstd[r] * g + b;
    }
  }
}

extern "C" void kernel_launch(void* const* d_in, const int* in_sizes, int n_in,
                              void* d_out, int out_size, void* d_ws, size_t ws_size,
                              hipStream_t stream) {
  const float* efeat    = (const float*)d_in[0];
  const float* src_feat = (const float*)d_in[1];
  const float* dst_feat = (const float*)d_in[2];
  const int*   src_idx  = (const int*)d_in[3];
  const int*   dst_idx  = (const int*)d_in[4];
  const float* w_efeat  = (const float*)d_in[5];
  const float* w_src    = (const float*)d_in[6];
  const float* w_dst    = (const float*)d_in[7];
  const float* b1       = (const float*)d_in[8];
  const float* w2       = (const float*)d_in[9];
  const float* b2       = (const float*)d_in[10];
  const float* ln_g     = (const float*)d_in[11];
  const float* ln_b     = (const float*)d_in[12];
  float* out = (float*)d_out;
  char* ws = (char*)d_ws;
  (void)n_in; (void)out_size; (void)ws_size;

  const long long E = (long long)in_sizes[0] / HID;

  // 1) split weights into WMMA-layout bf16 hi/lo fragments (256 KB of ws)
  prep_weights_kernel<<<KT1 * NT_TILES + KT2 * NT_TILES, 32, 0, stream>>>(
      w_efeat, w_src, w_dst, w2, ws);

  // 2) fused gather + GEMM1 + SiLU + GEMM2 + LayerNorm
  const long long tiles = (E + 15) / 16;
  const int grid = (int)((tiles + WAVES_PER_BLOCK - 1) / WAVES_PER_BLOCK);
  fused_edge_kernel<<<grid, WAVES_PER_BLOCK * 32, 0, stream>>>(
      efeat, src_feat, dst_feat, src_idx, dst_idx, b1, b2, ln_g, ln_b,
      ws, out, E);
}
